// ContextSNN_34428457845385
// MI455X (gfx1250) — matrix operand order repaired and checked by reading, not compiled, not run on hardware
//
#include <hip/hip_runtime.h>

// ---------------------------------------------------------------------------
// SNN (snntorch Leaky, subtract reset) on gfx1250 using v_wmma_f32_16x16x32_f16
//   layer1: cur1^T = W1[HID,IN] * X^T[IN,B]   (A = W1 rows, B = X rows)
//   layer2: cur2^T = W2[OUT,HID] * spk1^T     (A = W2 rows, B = spk1 rows)
// Transposed formulation => every WMMA fragment load is contiguous 16B/lane.
// Layer-1 wave tile is 32(M) x 64(N): 12 b128 loads per 8 WMMAs.
// ---------------------------------------------------------------------------

typedef __attribute__((ext_vector_type(16))) _Float16 v16h;
typedef __attribute__((ext_vector_type(8)))  _Float16 v8h;
typedef __attribute__((ext_vector_type(8)))  float    v8f;
typedef __attribute__((ext_vector_type(4)))  float    v4f;

namespace snn {
constexpr int T = 25, BATCH = 1024, IN = 1024, HID = 4096, OUT = 64;
constexpr float BETA = 0.9f;
}

// A fragment (16x32 f16, M x K): lane l -> row (l&15); halves hold
// K = 8*(l>>4)+0..7 and K = 16+8*(l>>4)+0..7  (ISA 7.12.2, 16-bit A 16x32)
__device__ __forceinline__ v16h load_a_frag(const _Float16* __restrict__ base,
                                            int lda, int row_base, int k0) {
  const int lane = threadIdx.x & 31;
  const int r = lane & 15, h = lane >> 4;
  const _Float16* p = base + (size_t)(row_base + r) * lda + (size_t)(k0 + 8 * h);
  v8h lo = *(const v8h*)(p);
  v8h hi = *(const v8h*)(p + 16);
  return __builtin_shufflevector(lo, hi, 0, 1, 2, 3, 4, 5, 6, 7,
                                 8, 9, 10, 11, 12, 13, 14, 15);
}

// B fragment (32x16 f16, K x N): lane l -> column (l&15);
// halves hold K = 8*(l>>4)+0..7 and K = 16+8*(l>>4)+0..7. Source rows (row-major [N][K]) act as
// columns of the transposed operand -> contiguous loads.
__device__ __forceinline__ v16h load_b_frag(const _Float16* __restrict__ base,
                                            int ldb, int col_base, int k0) {
  const int lane = threadIdx.x & 31;
  const int c = lane & 15, h = lane >> 4;
  const _Float16* p = base + (size_t)(col_base + c) * ldb + (size_t)(k0 + 8 * h);
  v8h lo = *(const v8h*)(p);
  v8h hi = *(const v8h*)(p + 16);
  return __builtin_shufflevector(lo, hi, 0, 1, 2, 3, 4, 5, 6, 7,
                                 8, 9, 10, 11, 12, 13, 14, 15);
}

// Fused Leaky-Integrate-Fire on one D tile column strip.
// D layout (ISA 7.12.2): lane l, VGPR v -> (M = v + 8*(l>>4), N = l&15).
// Here M = hidden index (contiguous per lane across v), N = batch row.
__device__ __forceinline__ void lif_update1(const v8f& acc, int brow, int hid0,
                                            const v4f& bl, const v4f& bh,
                                            float* __restrict__ mem1,
                                            _Float16* __restrict__ spk1) {
  const size_t idx = (size_t)brow * snn::HID + (size_t)hid0;
  float* mp = mem1 + idx;
  _Float16* sp = spk1 + idx;
  v4f m_lo = *(const v4f*)(mp);
  v4f m_hi = *(const v4f*)(mp + 4);
  v8h s_prev = *(const v8h*)(sp);
  v4f r_lo, r_hi;
  v8h s_new;
#pragma unroll
  for (int v = 0; v < 4; ++v) {
    float cur = acc[v] + bl[v];
    float m = snn::BETA * m_lo[v] + cur - (float)s_prev[v];
    r_lo[v] = m;
    s_new[v] = (m > 1.0f) ? (_Float16)1.0f : (_Float16)0.0f;
  }
#pragma unroll
  for (int v = 0; v < 4; ++v) {
    float cur = acc[4 + v] + bh[v];
    float m = snn::BETA * m_hi[v] + cur - (float)s_prev[4 + v];
    r_hi[v] = m;
    s_new[4 + v] = (m > 1.0f) ? (_Float16)1.0f : (_Float16)0.0f;
  }
  *(v4f*)(mp) = r_lo;
  *(v4f*)(mp + 4) = r_hi;
  *(v8h*)(sp) = s_new;
}

// Layer 1: each wave computes a 32(HID) x 64(BATCH) tile of cur1^T
// (8 WMMA accumulators, B fragments reused by two A fragments),
// then applies the LIF update in-register. 2048 waves total.
__global__ __launch_bounds__(256) void snn_layer1_kernel(
    const _Float16* __restrict__ xt,   // [BATCH][IN]  f16 spikes at time t
    const _Float16* __restrict__ w1f,  // [HID][IN]    f16
    const float* __restrict__ b1,      // [HID]
    float* __restrict__ mem1,          // [BATCH][HID]
    _Float16* __restrict__ spk1) {     // [BATCH][HID]
  using namespace snn;
  const int wave = blockIdx.x * (blockDim.x >> 5) + (threadIdx.x >> 5);
  const int m_base = (wave >> 4) << 5;  // hidden tile base  (HID/32 = 128)
  const int n_base = (wave & 15) << 6;  // batch macro-tile  (BATCH/64 = 16)

  v8f acc[8];
#pragma unroll
  for (int j = 0; j < 8; ++j) acc[j] = (v8f){};

  for (int k0 = 0; k0 < IN; k0 += 32) {
    v16h a0 = load_a_frag(w1f, IN, m_base, k0);
    v16h a1 = load_a_frag(w1f, IN, m_base + 16, k0);
#pragma unroll
    for (int j = 0; j < 4; ++j) {
      v16h b = load_b_frag(xt, IN, n_base + 16 * j, k0);
      acc[j] = __builtin_amdgcn_wmma_f32_16x16x32_f16(
          false, a0, false, b, (short)0, acc[j], false, false);
      acc[4 + j] = __builtin_amdgcn_wmma_f32_16x16x32_f16(
          false, a1, false, b, (short)0, acc[4 + j], false, false);
    }
  }

  const int lane = threadIdx.x & 31;
  const int hid0a = m_base + ((lane >> 4) << 3);
  const int hid0b = hid0a + 16;
  const int bn = lane & 15;
  v4f ba_lo = *(const v4f*)(b1 + hid0a);
  v4f ba_hi = *(const v4f*)(b1 + hid0a + 4);
  v4f bb_lo = *(const v4f*)(b1 + hid0b);
  v4f bb_hi = *(const v4f*)(b1 + hid0b + 4);
#pragma unroll
  for (int j = 0; j < 4; ++j) {
    const int brow = n_base + 16 * j + bn;
    lif_update1(acc[j], brow, hid0a, ba_lo, ba_hi, mem1, spk1);
    lif_update1(acc[4 + j], brow, hid0b, bb_lo, bb_hi, mem1, spk1);
  }
}

// Layer 2: each wave computes one 16(OUT) x 16(BATCH) tile of cur2^T over the
// full K = 4096, applies LIF, and accumulates spikes into the output.
// 256 waves, 128 WMMAs each (layer 1 issues 256 per wave).
__global__ __launch_bounds__(256) void snn_layer2_kernel(
    const _Float16* __restrict__ spk1,  // [BATCH][HID] f16
    const _Float16* __restrict__ w2f,   // [OUT][HID]   f16
    const float* __restrict__ b2,       // [OUT]
    float* __restrict__ mem2,           // [BATCH][OUT]
    float* __restrict__ spk2,           // [BATCH][OUT]
    float* __restrict__ out) {          // [BATCH][OUT] spike counts
  using namespace snn;
  const int wave = blockIdx.x * (blockDim.x >> 5) + (threadIdx.x >> 5);
  const int m_base = (wave & 3) << 4;   // OUT/16 = 4
  const int n_base = (wave >> 2) << 4;  // BATCH/16 = 64

  v8f acc = {};
  for (int k0 = 0; k0 < HID; k0 += 32) {
    v16h a = load_a_frag(w2f, HID, m_base, k0);
    v16h b = load_b_frag(spk1, HID, n_base, k0);
    acc = __builtin_amdgcn_wmma_f32_16x16x32_f16(false, a, false, b,
                                                 (short)0, acc, false, false);
  }

  const int lane = threadIdx.x & 31;
  const int brow = n_base + (lane & 15);
  const int o0 = m_base + ((lane >> 4) << 3);
  const size_t idx = (size_t)brow * OUT + (size_t)o0;
  v4f bias_lo = *(const v4f*)(b2 + o0);
  v4f bias_hi = *(const v4f*)(b2 + o0 + 4);
  v4f m_lo = *(const v4f*)(mem2 + idx);
  v4f m_hi = *(const v4f*)(mem2 + idx + 4);
  v4f s_lo = *(const v4f*)(spk2 + idx);
  v4f s_hi = *(const v4f*)(spk2 + idx + 4);
  v4f o_lo = *(const v4f*)(out + idx);
  v4f o_hi = *(const v4f*)(out + idx + 4);
#pragma unroll
  for (int v = 0; v < 4; ++v) {
    float m = snn::BETA * m_lo[v] + (acc[v] + bias_lo[v]) - s_lo[v];
    float s = (m > 1.0f) ? 1.0f : 0.0f;
    m_lo[v] = m; s_lo[v] = s; o_lo[v] += s;
  }
#pragma unroll
  for (int v = 0; v < 4; ++v) {
    float m = snn::BETA * m_hi[v] + (acc[4 + v] + bias_hi[v]) - s_hi[v];
    float s = (m > 1.0f) ? 1.0f : 0.0f;
    m_hi[v] = m; s_hi[v] = s; o_hi[v] += s;
  }
  *(v4f*)(mem2 + idx) = m_lo;
  *(v4f*)(mem2 + idx + 4) = m_hi;
  *(v4f*)(spk2 + idx) = s_lo;
  *(v4f*)(spk2 + idx + 4) = s_hi;
  *(v4f*)(out + idx) = o_lo;
  *(v4f*)(out + idx + 4) = o_hi;
}

__global__ __launch_bounds__(256) void snn_f32_to_f16_kernel(
    const float* __restrict__ src, _Float16* __restrict__ dst, size_t n) {
  size_t i = (size_t)blockIdx.x * blockDim.x + threadIdx.x;
  const size_t stride = (size_t)gridDim.x * blockDim.x;
  for (; i < n; i += stride) dst[i] = (_Float16)src[i];
}

__global__ __launch_bounds__(256) void snn_zero_u32_kernel(
    unsigned* __restrict__ p, size_t nwords) {
  size_t i = (size_t)blockIdx.x * blockDim.x + threadIdx.x;
  const size_t stride = (size_t)gridDim.x * blockDim.x;
  for (; i < nwords; i += stride) p[i] = 0u;
}

extern "C" void kernel_launch(void* const* d_in, const int* in_sizes, int n_in,
                              void* d_out, int out_size, void* d_ws,
                              size_t ws_size, hipStream_t stream) {
  using namespace snn;
  const float* spike_seq = (const float*)d_in[0];  // [T][BATCH][IN]
  const float* w1 = (const float*)d_in[1];         // [HID][IN]
  const float* b1 = (const float*)d_in[2];         // [HID]
  const float* w2 = (const float*)d_in[3];         // [OUT][HID]
  const float* b2 = (const float*)d_in[4];         // [OUT]
  float* out = (float*)d_out;                      // [BATCH][OUT]

  // Workspace carve-up (all region sizes are multiples of 256 bytes).
  char* ws = (char*)d_ws;
  _Float16* xf16 = (_Float16*)ws; ws += (size_t)T * BATCH * IN * sizeof(_Float16);
  _Float16* w1f  = (_Float16*)ws; ws += (size_t)HID * IN * sizeof(_Float16);
  _Float16* w2f  = (_Float16*)ws; ws += (size_t)OUT * HID * sizeof(_Float16);
  float*    mem1 = (float*)ws;    ws += (size_t)BATCH * HID * sizeof(float);
  _Float16* spk1 = (_Float16*)ws; ws += (size_t)BATCH * HID * sizeof(_Float16);
  float*    mem2 = (float*)ws;    ws += (size_t)BATCH * OUT * sizeof(float);
  float*    spk2 = (float*)ws;    ws += (size_t)BATCH * OUT * sizeof(float);

  // Zero the recurrent state (mem1/spk1/mem2/spk2 are contiguous) and output.
  const size_t state_words =
      ((size_t)BATCH * HID * sizeof(float) +
       (size_t)BATCH * HID * sizeof(_Float16) +
       2 * (size_t)BATCH * OUT * sizeof(float)) / 4;
  snn_zero_u32_kernel<<<1024, 256, 0, stream>>>((unsigned*)mem1, state_words);
  snn_zero_u32_kernel<<<64, 256, 0, stream>>>((unsigned*)out,
                                              (size_t)BATCH * OUT);

  // One-shot f32 -> f16 conversions (spikes are exactly representable).
  snn_f32_to_f16_kernel<<<2048, 256, 0, stream>>>(spike_seq, xf16,
                                                  (size_t)T * BATCH * IN);
  snn_f32_to_f16_kernel<<<512, 256, 0, stream>>>(w1, w1f, (size_t)HID * IN);
  snn_f32_to_f16_kernel<<<64, 256, 0, stream>>>(w2, w2f, (size_t)OUT * HID);

  // Sequential scan over time: layer1 GEMM+LIF, then layer2 GEMM+LIF+count.
  for (int t = 0; t < T; ++t) {
    const _Float16* xt = xf16 + (size_t)t * BATCH * IN;
    snn_layer1_kernel<<<256, 256, 0, stream>>>(xt, w1f, b1, mem1, spk1);
    snn_layer2_kernel<<<32, 256, 0, stream>>>(spk1, w2f, b2, mem2, spk2, out);
  }
}
